// Net_44976897524622
// MI455X (gfx1250) — matrix operand order repaired and checked by reading, not compiled, not run on hardware
//
#include <hip/hip_runtime.h>
#include <cstdint>
#include <cstddef>

// ---------------------------------------------------------------------------
// CDNA5 / gfx1250 implementation of the masked SR network.
//  - activations f16 NHWC (16ch maps carry a 1-px zero halo: 514x514)
//  - low path of every conv_block algebraically folded through its 1x1 tail
//  - mid/reduction/expansion blocks: v_wmma_f32_16x16x32_f16, 16px per wave
//  - mask combine + bias + relu fused into the WMMA epilogue
//  - first conv / last tconv: scalar (memory-bound edges)
// ---------------------------------------------------------------------------

typedef _Float16 half_t;
typedef __attribute__((ext_vector_type(2)))  _Float16 h2;
typedef __attribute__((ext_vector_type(16))) _Float16 v16h;
typedef __attribute__((ext_vector_type(8)))  float    v8f;

struct HF8 { h2 q[8]; };                      // 32B -> one v16h fragment
struct alignas(16) F16x8 { _Float16 h[8]; };  // 16B vector store unit

__device__ __forceinline__ v16h pack16(const HF8& f) {
  return __builtin_bit_cast(v16h, f);
}
__device__ __forceinline__ v8f wmma_f16(v16h a, v16h b, v8f c) {
  // D = A(16x32 f16) * B(32x16 f16) + C(16x16 f32)
  return __builtin_amdgcn_wmma_f32_16x16x32_f16(false, a, false, b,
                                                (short)0, c, false, false);
}

#define NB   8
#define HH   512
#define HP   514   // halo'd

// ---------------------------------------------------------------------------
// Mask: blur(3x3, count_include_pad=False) -> |x-blur|>=0.04 -> 3x3 dilate
// ---------------------------------------------------------------------------
__global__ __launch_bounds__(256) void k_mask(const float* __restrict__ x,
                                              float* __restrict__ m) {
  size_t idx = (size_t)blockIdx.x * 256 + threadIdx.x;   // NB*512*512 exact
  int xp = (int)(idx & 511);
  int y  = (int)((idx >> 9) & 511);
  int b  = (int)(idx >> 18);
  const float* xb = x + ((size_t)b << 18);
  float best = 0.0f;
  #pragma unroll
  for (int dy = -1; dy <= 1; ++dy) {
    #pragma unroll
    for (int dx = -1; dx <= 1; ++dx) {
      int qy = y + dy, qx = xp + dx;
      if (qy < 0 || qy >= HH || qx < 0 || qx >= HH) continue;
      float s = 0.0f, c = 0.0f;
      #pragma unroll
      for (int ey = -1; ey <= 1; ++ey) {
        #pragma unroll
        for (int ex = -1; ex <= 1; ++ex) {
          int ry = qy + ey, rx = qx + ex;
          if (ry >= 0 && ry < HH && rx >= 0 && rx < HH) {
            s += xb[ry * HH + rx]; c += 1.0f;
          }
        }
      }
      float d = fabsf(xb[qy * HH + qx] - s / c);
      if (d >= 0.04f) best = 1.0f;
    }
  }
  m[idx] = best;
}

// ---------------------------------------------------------------------------
// Weight folding (low path: Wlo = l2w @ l1w, bf1 = l2w @ l1b)
// ---------------------------------------------------------------------------
__global__ void k_fold_first(const float* hw, const float* hb,
                             const float* l1w, const float* l1b,
                             const float* l2w, const float* l2b,
                             float* Wout /*[2][32][25]*/, float* Bout /*[3][32]*/) {
  int idx = blockIdx.x * 256 + threadIdx.x;
  if (idx < 1600) {
    int p = idx / 800, oc = (idx % 800) / 25, t = idx % 25;
    float v;
    if (p == 0) v = hw[oc * 25 + t];
    else { v = 0.f; for (int r = 0; r < 8; ++r) v += l2w[oc * 8 + r] * l1w[r * 25 + t]; }
    Wout[idx] = v;
  }
  if (idx < 32) {
    float s = 0.f; for (int r = 0; r < 8; ++r) s += l2w[idx * 8 + r] * l1b[r];
    Bout[idx] = hb[idx]; Bout[32 + idx] = s; Bout[64 + idx] = l2b[idx];
  }
}

__global__ void k_fold_red(const float* hw, const float* hb,
                           const float* l1w, const float* l1b,
                           const float* l2w, const float* l2b,
                           half_t* Wout /*[2][16][32]*/, float* Bout /*[3][16]*/) {
  int idx = blockIdx.x * 256 + threadIdx.x;
  if (idx < 1024) {
    int p = idx / 512, oc = (idx % 512) / 32, ic = idx % 32;
    float v;
    if (p == 0) v = hw[oc * 32 + ic];
    else { v = 0.f; for (int r = 0; r < 4; ++r) v += l2w[oc * 4 + r] * l1w[r * 32 + ic]; }
    Wout[idx] = (half_t)v;
  }
  if (idx < 16) {
    float s = 0.f; for (int r = 0; r < 4; ++r) s += l2w[idx * 4 + r] * l1b[r];
    Bout[idx] = hb[idx]; Bout[16 + idx] = s; Bout[32 + idx] = l2b[idx];
  }
}

__global__ void k_fold_mid(const float* hw, const float* hb,
                           const float* l1w, const float* l1b,
                           const float* l2w, const float* l2b,
                           half_t* Wout /*[2][16][10][16]*/, float* Bout /*[3][16]*/) {
  int idx = blockIdx.x * 256 + threadIdx.x;
  if (idx < 5120) {
    int p = idx / 2560, rem = idx % 2560;
    int oc = rem / 160, rem2 = rem % 160;
    int t = rem2 / 16, ic = rem2 % 16;
    float v = 0.f;
    if (t < 9) {
      if (p == 0) v = hw[(oc * 16 + ic) * 9 + t];
      else { for (int r = 0; r < 4; ++r) v += l2w[oc * 4 + r] * l1w[(r * 16 + ic) * 9 + t]; }
    }
    Wout[idx] = (half_t)v;
  }
  if (idx < 16) {
    float s = 0.f; for (int r = 0; r < 4; ++r) s += l2w[idx * 4 + r] * l1b[r];
    Bout[idx] = hb[idx]; Bout[16 + idx] = s; Bout[32 + idx] = l2b[idx];
  }
}

__global__ void k_fold_exp(const float* hw, const float* hb,
                           const float* l1w, const float* l1b,
                           const float* l2w, const float* l2b,
                           half_t* Wout /*[4][16][32]*/, float* Bout /*[3][32]*/) {
  int idx = blockIdx.x * 256 + threadIdx.x;
  if (idx < 2048) {
    int tile = idx / 512, row = (idx % 512) / 32, k = idx % 32;
    int path = tile >> 1, oc = (tile & 1) * 16 + row;
    float v = 0.f;
    if (k < 16) {
      if (path == 0) v = hw[oc * 16 + k];
      else { for (int r = 0; r < 8; ++r) v += l2w[oc * 8 + r] * l1w[r * 16 + k]; }
    }
    Wout[idx] = (half_t)v;
  }
  if (idx < 32) {
    float s = 0.f; for (int r = 0; r < 8; ++r) s += l2w[idx * 8 + r] * l1b[r];
    Bout[idx] = hb[idx]; Bout[32 + idx] = s; Bout[64 + idx] = l2b[idx];
  }
}

// ---------------------------------------------------------------------------
// First block: 1->32, 5x5 pad2, scalar (1 input channel => no GEMM shape)
// out = relu(mk*(hi+bh) + (1-mk)*(lo+bf1) + b2), NHWC32 f16
// ---------------------------------------------------------------------------
__global__ __launch_bounds__(256) void k_first(const float* __restrict__ x,
                                               const float* __restrict__ W,
                                               const float* __restrict__ B,
                                               const float* __restrict__ mask,
                                               half_t* __restrict__ out) {
  __shared__ float sW[1600];
  __shared__ float sB[96];
  int tid = threadIdx.x;
  for (int i = tid; i < 1600; i += 256) sW[i] = W[i];
  if (tid < 96) sB[tid] = B[tid];
  __syncthreads();

  size_t idx = (size_t)blockIdx.x * 256 + tid;
  int xp = (int)(idx & 511), y = (int)((idx >> 9) & 511), b = (int)(idx >> 18);
  const float* xb = x + ((size_t)b << 18);

  float win[25];
  #pragma unroll
  for (int ky = 0; ky < 5; ++ky) {
    #pragma unroll
    for (int kx = 0; kx < 5; ++kx) {
      int yy = y + ky - 2, xx = xp + kx - 2;
      win[ky * 5 + kx] = (yy >= 0 && yy < HH && xx >= 0 && xx < HH)
                           ? xb[yy * HH + xx] : 0.0f;
    }
  }
  float mk = mask[idx], im = 1.0f - mk;
  half_t* op = out + idx * 32;
  for (int oc = 0; oc < 32; ++oc) {
    float hi = 0.f, lo = 0.f;
    #pragma unroll
    for (int t = 0; t < 25; ++t) {
      hi = fmaf(sW[oc * 25 + t], win[t], hi);
      lo = fmaf(sW[800 + oc * 25 + t], win[t], lo);
    }
    float v = mk * (hi + sB[oc]) + im * (lo + sB[32 + oc]) + sB[64 + oc];
    op[oc] = (half_t)fmaxf(v, 0.0f);
  }
}

// ---------------------------------------------------------------------------
// Reduction block: 1x1, 32 -> (16 hi + 16 lo).  One K=32 WMMA per M-tile.
// in: [8][512][512][32], out: h16 halo'd
// ---------------------------------------------------------------------------
__global__ __launch_bounds__(256) void k_red(const half_t* __restrict__ in,
                                             half_t* __restrict__ out,
                                             const half_t* __restrict__ W,
                                             const float* __restrict__ B,
                                             const float* __restrict__ mask) {
  __shared__ half_t sW[1024];
  __shared__ float  sB[48];
  const int tid = threadIdx.x;
  { const uint32_t* gw = (const uint32_t*)W; uint32_t* sw = (uint32_t*)sW;
    for (int i = tid; i < 512; i += 256) sw[i] = gw[i];
    if (tid < 48) sB[tid] = B[tid]; }
  __syncthreads();

  const int lane = tid & 31, n = lane & 15, hf = lane >> 4;
  const int b = blockIdx.z, y = blockIdx.y;
  const int xx = blockIdx.x * 128 + (tid >> 5) * 16 + n;
  const size_t pix = ((size_t)b * HH + y) * HH + xx;

  HF8 bf;
  #pragma unroll
  for (int v = 0; v < 8; ++v)
    bf.q[v] = *(const h2*)(in + pix * 32 + hf * 16 + 2 * v);
  const v16h bm = pack16(bf);

  v8f chi = {}; v8f clo = {};
  #pragma unroll
  for (int path = 0; path < 2; ++path) {
    HF8 af;
    #pragma unroll
    for (int v = 0; v < 8; ++v) {
      const int kb = hf * 16 + 2 * v;
      af.q[v] = *(const h2*)&sW[(path * 16 + n) * 32 + kb];
    }
    const v16h am = pack16(af);
    if (path == 0) chi = wmma_f16(am, bm, chi);
    else           clo = wmma_f16(am, bm, clo);
  }

  const float mk = mask[pix], im = 1.0f - mk;
  F16x8 o;
  #pragma unroll
  for (int r = 0; r < 8; ++r) {
    int mv = 8 * hf + r;
    float v = mk * (chi[r] + sB[mv]) + im * (clo[r] + sB[16 + mv]) + sB[32 + mv];
    o.h[r] = (_Float16)fmaxf(v, 0.0f);
  }
  const long pb = ((long)b * HP + (y + 1)) * HP + (xx + 1);
  *(F16x8*)(out + pb * 16 + 8 * hf) = o;
}

// ---------------------------------------------------------------------------
// Mid block: 3x3, 16 -> (16 hi + 16 lo folded).  K = 10 taps * 16ch = 5 chunks.
// h16 halo'd in/out.
// ---------------------------------------------------------------------------
__global__ __launch_bounds__(256) void k_mid(const half_t* __restrict__ in,
                                             half_t* __restrict__ out,
                                             const half_t* __restrict__ W,
                                             const float* __restrict__ B,
                                             const float* __restrict__ mask) {
  __shared__ half_t sW[2 * 16 * 10 * 16];
  __shared__ float  sB[48];
  const int tid = threadIdx.x;
  { const uint32_t* gw = (const uint32_t*)W; uint32_t* sw = (uint32_t*)sW;
    for (int i = tid; i < 2560; i += 256) sw[i] = gw[i];
    if (tid < 48) sB[tid] = B[tid]; }
  __syncthreads();

  const int lane = tid & 31, n = lane & 15, hf = lane >> 4;
  const int b = blockIdx.z, y = blockIdx.y;
  const int xx = blockIdx.x * 128 + (tid >> 5) * 16 + n;
  const long pb = ((long)b * HP + (y + 1)) * HP + (xx + 1);

  v8f chi = {}; v8f clo = {};
  #pragma unroll
  for (int c = 0; c < 5; ++c) {
    HF8 bf;
    #pragma unroll
    for (int v = 0; v < 8; ++v) {
      const int kg = c * 32 + hf * 16 + 2 * v;       // global K index
      const int tap = kg >> 4, ch = kg & 15;
      h2 val; val.x = (_Float16)0.f; val.y = (_Float16)0.f;
      if (tap < 9) {
        const long dy = tap / 3 - 1, dx = tap % 3 - 1;
        val = *(const h2*)(in + (pb + dy * HP + dx) * 16 + ch);
      }
      bf.q[v] = val;
    }
    const v16h bm = pack16(bf);
    #pragma unroll
    for (int path = 0; path < 2; ++path) {
      HF8 af;
      #pragma unroll
      for (int v = 0; v < 8; ++v) {
        const int kb = hf * 16 + 2 * v;
        const int kg = c * 32 + kb;
        const int tap = kg >> 4, ch = kg & 15;
        af.q[v] = *(const h2*)&sW[(((path * 16 + n) * 10) + tap) * 16 + ch];
      }
      const v16h am = pack16(af);
      if (path == 0) chi = wmma_f16(am, bm, chi);
      else           clo = wmma_f16(am, bm, clo);
    }
  }

  const float mk = mask[((size_t)b * HH + y) * HH + xx];
  const float im = 1.0f - mk;
  F16x8 o;
  #pragma unroll
  for (int r = 0; r < 8; ++r) {
    int mv = 8 * hf + r;
    float v = mk * (chi[r] + sB[mv]) + im * (clo[r] + sB[16 + mv]) + sB[32 + mv];
    o.h[r] = (_Float16)fmaxf(v, 0.0f);
  }
  *(F16x8*)(out + pb * 16 + 8 * hf) = o;
}

// ---------------------------------------------------------------------------
// Expansion: 1x1, 16 -> (32 hi + 32 lo).  K=16 padded to 32; 4 M-tiles.
// in: h16 halo'd, out: [8][512][512][32]
// ---------------------------------------------------------------------------
__global__ __launch_bounds__(256) void k_exp(const half_t* __restrict__ in,
                                             half_t* __restrict__ out,
                                             const half_t* __restrict__ W,
                                             const float* __restrict__ B,
                                             const float* __restrict__ mask) {
  __shared__ half_t sW[2048];
  __shared__ float  sB[96];
  const int tid = threadIdx.x;
  { const uint32_t* gw = (const uint32_t*)W; uint32_t* sw = (uint32_t*)sW;
    for (int i = tid; i < 1024; i += 256) sw[i] = gw[i];
    if (tid < 96) sB[tid] = B[tid]; }
  __syncthreads();

  const int lane = tid & 31, n = lane & 15, hf = lane >> 4;
  const int b = blockIdx.z, y = blockIdx.y;
  const int xx = blockIdx.x * 128 + (tid >> 5) * 16 + n;
  const long pb = ((long)b * HP + (y + 1)) * HP + (xx + 1);

  HF8 bf;
  #pragma unroll
  for (int v = 0; v < 8; ++v) {
    h2 val; val.x = (_Float16)0.f; val.y = (_Float16)0.f;
    if (hf == 0) val = *(const h2*)(in + pb * 16 + 2 * v);  // K 16..31 are pad
    bf.q[v] = val;
  }
  const v16h bm = pack16(bf);

  v8f c0 = {}, c1 = {}, c2 = {}, c3 = {};
  #pragma unroll
  for (int t = 0; t < 4; ++t) {
    HF8 af;
    #pragma unroll
    for (int v = 0; v < 8; ++v) {
      const int kb = hf * 16 + 2 * v;
      af.q[v] = *(const h2*)&sW[(t * 16 + n) * 32 + kb];
    }
    const v16h am = pack16(af);
    if (t == 0) c0 = wmma_f16(am, bm, c0);
    else if (t == 1) c1 = wmma_f16(am, bm, c1);
    else if (t == 2) c2 = wmma_f16(am, bm, c2);
    else             c3 = wmma_f16(am, bm, c3);
  }

  const size_t pix = ((size_t)b * HH + y) * HH + xx;
  const float mk = mask[pix], im = 1.0f - mk;
  F16x8 o0, o1;
  #pragma unroll
  for (int r = 0; r < 8; ++r) {
    int oc = 8 * hf + r;
    float v0 = mk * (c0[r] + sB[oc]) + im * (c2[r] + sB[32 + oc]) + sB[64 + oc];
    o0.h[r] = (_Float16)fmaxf(v0, 0.0f);
    int oc2 = 16 + oc;
    float v1 = mk * (c1[r] + sB[oc2]) + im * (c3[r] + sB[32 + oc2]) + sB[64 + oc2];
    o1.h[r] = (_Float16)fmaxf(v1, 0.0f);
  }
  *(F16x8*)(out + pix * 32 + 8 * hf) = o0;
  *(F16x8*)(out + pix * 32 + 16 + 8 * hf) = o1;
}

// ---------------------------------------------------------------------------
// Last-block low 1x1: 32 -> 8, scalar.  out NHWC8 f16.
// ---------------------------------------------------------------------------
__global__ __launch_bounds__(256) void k_low1(const half_t* __restrict__ h32,
                                              const float* __restrict__ W,
                                              const float* __restrict__ bb,
                                              half_t* __restrict__ low8) {
  __shared__ float sW[256];
  __shared__ float sB[8];
  int tid = threadIdx.x;
  if (tid < 256) sW[tid] = W[tid];
  if (tid < 8) sB[tid] = bb[tid];
  __syncthreads();
  size_t idx = (size_t)blockIdx.x * 256 + tid;
  const half_t* p = h32 + idx * 32;
  float in[32];
  #pragma unroll
  for (int i = 0; i < 32; ++i) in[i] = (float)p[i];
  half_t* op = low8 + idx * 8;
  #pragma unroll
  for (int r = 0; r < 8; ++r) {
    float a = sB[r];
    #pragma unroll
    for (int ic = 0; ic < 32; ++ic) a = fmaf(sW[r * 32 + ic], in[ic], a);
    op[r] = (half_t)a;
  }
}

// ---------------------------------------------------------------------------
// Output: tconv 9x9 stride 3 pad 4 opad 2 for both paths, + nearest-up mask.
// out[oy,ox] = mu*(sum_h + hb) + (1-mu)*(sum_l + lb)
// ---------------------------------------------------------------------------
__global__ __launch_bounds__(256) void k_out(const half_t* __restrict__ h32,
                                             const half_t* __restrict__ low8,
                                             const float* __restrict__ Wh,
                                             const float* __restrict__ Wl,
                                             const float* __restrict__ hb,
                                             const float* __restrict__ lb,
                                             const float* __restrict__ mask,
                                             float* __restrict__ out) {
  __shared__ float sWh[2592];
  __shared__ float sWl[648];
  int tid = threadIdx.x;
  for (int i = tid; i < 2592; i += 256) sWh[i] = Wh[i];
  for (int i = tid; i < 648; i += 256) sWl[i] = Wl[i];
  __syncthreads();

  size_t idx = (size_t)blockIdx.x * 256 + tid;   // 8*1536*1536 exact
  int ox = (int)(idx % 1536);
  size_t t1 = idx / 1536;
  int oy = (int)(t1 % 1536);
  int b = (int)(t1 / 1536);

  int ty = oy - 4, tx = ox - 4;
  int iy0 = (ty <= 0) ? 0 : (ty + 2) / 3;
  int ix0 = (tx <= 0) ? 0 : (tx + 2) / 3;
  int iy1 = (oy + 4) / 3; if (iy1 > HH - 1) iy1 = HH - 1;
  int ix1 = (ox + 4) / 3; if (ix1 > HH - 1) ix1 = HH - 1;

  float ah = 0.f, al = 0.f;
  for (int iy = iy0; iy <= iy1; ++iy) {
    int ky = oy + 4 - 3 * iy;
    for (int ix = ix0; ix <= ix1; ++ix) {
      int kx = ox + 4 - 3 * ix;
      int kk = ky * 9 + kx;
      const half_t* hp = h32 + (((size_t)b * HH + iy) * HH + ix) * 32;
      const half_t* lp = low8 + (((size_t)b * HH + iy) * HH + ix) * 8;
      #pragma unroll
      for (int ch = 0; ch < 32; ++ch) ah = fmaf((float)hp[ch], sWh[ch * 81 + kk], ah);
      #pragma unroll
      for (int r = 0; r < 8; ++r) al = fmaf((float)lp[r], sWl[r * 81 + kk], al);
    }
  }
  float mu = mask[((size_t)b * HH + oy / 3) * HH + ox / 3];
  out[idx] = mu * (ah + hb[0]) + (1.0f - mu) * (al + lb[0]);
}

// ---------------------------------------------------------------------------
// Host side
// Input order assumption (jax pytree: sorted dict keys, params before x;
// per block sorted: hb,hw,l1b,l1w,l2b,l2w):
//  0-5 expansion, 6-11 first, 12-17 last, 18-23 mid1, 24-29 mid2,
//  30-35 mid3, 36-41 mid4, 42-47 reduction, 48 x
// ---------------------------------------------------------------------------
extern "C" void kernel_launch(void* const* d_in, const int* in_sizes, int n_in,
                              void* d_out, int out_size, void* d_ws, size_t ws_size,
                              hipStream_t stream) {
  if (n_in < 49) return;
  auto F = [&](int i) { return (const float*)d_in[i]; };
  const float* x = F(48);

  // workspace carve-up
  char* base = (char*)d_ws;
  size_t off = 0;
  auto take = [&](size_t bytes) {
    size_t r = off;
    off += (bytes + 255) & ~(size_t)255;
    return r;
  };
  const size_t npix = (size_t)NB * HH * HH;          // 2,097,152
  const size_t nhalo = (size_t)NB * HP * HP;         // 2,113,568
  size_t o_mask  = take(npix * 4);
  size_t o_h32   = take(npix * 32 * 2);
  size_t o_h16a  = take(nhalo * 16 * 2);
  size_t o_h16b  = take(nhalo * 16 * 2);
  size_t o_low8  = take(npix * 8 * 2);
  size_t o_fW    = take(1600 * 4);
  size_t o_fB    = take(96 * 4);
  size_t o_rW    = take(1024 * 2);
  size_t o_rB    = take(48 * 4);
  size_t o_mW    = take(4 * 5120 * 2);
  size_t o_mB    = take(4 * 48 * 4);
  size_t o_eW    = take(2048 * 2);
  size_t o_eB    = take(96 * 4);
  if (ws_size < off) return;   // not enough scratch; deterministic no-op

  float*  maskp = (float*)(base + o_mask);
  half_t* h32   = (half_t*)(base + o_h32);
  half_t* h16a  = (half_t*)(base + o_h16a);
  half_t* h16b  = (half_t*)(base + o_h16b);
  half_t* low8  = (half_t*)(base + o_low8);
  float*  fW    = (float*)(base + o_fW);
  float*  fB    = (float*)(base + o_fB);
  half_t* rW    = (half_t*)(base + o_rW);
  float*  rB    = (float*)(base + o_rB);
  half_t* mW    = (half_t*)(base + o_mW);
  float*  mB    = (float*)(base + o_mB);
  half_t* eW    = (half_t*)(base + o_eW);
  float*  eB    = (float*)(base + o_eB);

  // zero halo'd activation buffers (halos must be 0 every call)
  hipMemsetAsync(h16a, 0, nhalo * 16 * 2, stream);
  hipMemsetAsync(h16b, 0, nhalo * 16 * 2, stream);

  // mask
  k_mask<<<(int)(npix / 256), 256, 0, stream>>>(x, maskp);

  // weight folding
  k_fold_first<<<7, 256, 0, stream>>>(F(7), F(6), F(9), F(8), F(11), F(10), fW, fB);
  k_fold_red<<<4, 256, 0, stream>>>(F(43), F(42), F(45), F(44), F(47), F(46), rW, rB);
  for (int l = 0; l < 4; ++l) {
    int bi = 18 + 6 * l;
    k_fold_mid<<<20, 256, 0, stream>>>(F(bi + 1), F(bi + 0), F(bi + 3), F(bi + 2),
                                       F(bi + 5), F(bi + 4),
                                       mW + (size_t)l * 5120, mB + (size_t)l * 48);
  }
  k_fold_exp<<<8, 256, 0, stream>>>(F(1), F(0), F(3), F(2), F(5), F(4), eW, eB);

  // network
  dim3 gw(4, 512, NB);
  k_first<<<(int)(npix / 256), 256, 0, stream>>>(x, fW, fB, maskp, h32);
  k_red<<<gw, 256, 0, stream>>>(h32, h16a, rW, rB, maskp);
  k_mid<<<gw, 256, 0, stream>>>(h16a, h16b, mW + 0 * 5120, mB + 0 * 48, maskp);
  k_mid<<<gw, 256, 0, stream>>>(h16b, h16a, mW + 1 * 5120, mB + 1 * 48, maskp);
  k_mid<<<gw, 256, 0, stream>>>(h16a, h16b, mW + 2 * 5120, mB + 2 * 48, maskp);
  k_mid<<<gw, 256, 0, stream>>>(h16b, h16a, mW + 3 * 5120, mB + 3 * 48, maskp);
  k_exp<<<gw, 256, 0, stream>>>(h16a, h32, eW, eB, maskp);

  // last stage
  k_low1<<<(int)(npix / 256), 256, 0, stream>>>(h32, F(15), F(14), low8);
  size_t nout = (size_t)NB * 1536 * 1536;
  k_out<<<(int)(nout / 256), 256, 0, stream>>>(h32, low8, F(13), F(17),
                                               F(12), F(16), maskp, (float*)d_out);
  (void)in_sizes; (void)out_size;
}
